// GlobalPool_46866683134576
// MI455X (gfx1250) — matrix operand order repaired and checked by reading, not compiled, not run on hardware
//
#include <hip/hip_runtime.h>
#include <hip/hip_bf16.h>

// ---------------------------------------------------------------------------
// MI455X (gfx1250) graph attention pooling + GRUCell.
//
// Roofline: with the softmax-weights-sum-to-1 identity,
//   context_g = elu( Wp @ (sum_v a_v * nf_v) + bp ),
// the V-sized projection GEMM disappears and the problem is HBM-bound on two
// streaming passes over node_feats (2 x 256 MB ~= 22us at 23.3 TB/s).
// Remaining GEMMs (~4 GFLOP) run on v_wmma_f32_16x16x32_bf16 with all
// operands pre-converted to bf16 once (packed b128 fragment loads, no cvt in
// the inner loops). The dominant streaming loop uses the gfx1250 async
// global->LDS DMA path (ASYNCcnt) for depth-4 pipelining when available.
// ---------------------------------------------------------------------------

#define NFEAT 256
#define GFEAT 256
#define NV    262144
#define NG    4096

typedef __attribute__((ext_vector_type(16))) __bf16 v16bf;
typedef __attribute__((ext_vector_type(8)))  __bf16 v8bf;
typedef __attribute__((ext_vector_type(4)))  __bf16 v4bf;
typedef __attribute__((ext_vector_type(8)))  float  v8f;

#if defined(__gfx1250__) && \
    __has_builtin(__builtin_amdgcn_global_load_async_to_lds_b32) && \
    __has_builtin(__builtin_amdgcn_s_wait_asynccnt)
#define USE_ASYNC_LDS 1
#else
#define USE_ASYNC_LDS 0
#endif

__device__ __forceinline__ float sigmoidf_(float x) {
    return 1.0f / (1.0f + __expf(-x));
}

__device__ __forceinline__ v8f wmma_bf16(v16bf a, v16bf b, v8f c) {
    // (neg_a, A, neg_b, B, c_mod, C, reuse_a, reuse_b)
    return __builtin_amdgcn_wmma_f32_16x16x32_bf16(false, a, false, b,
                                                   (short)0, c, false, false);
}

// A-matrix fragment (16x32, M x K), 16-bit layout per ISA 7.12.2:
// lane l: m = l&15; kb = (l<16)?0:8; element i: k = (i>>3)*16 + kb + (i&7).
// rowptr points at A_bf16[m_global][kk]; two packed 16B loads.
__device__ __forceinline__ v16bf load_a_frag(const __bf16* __restrict__ rowptr, int lane) {
    const int kb = (lane & 16) ? 8 : 0;
    v8bf q0 = *reinterpret_cast<const v8bf*>(rowptr + kb);        // k = kb..kb+7
    v8bf q1 = *reinterpret_cast<const v8bf*>(rowptr + kb + 16);   // k = 16+kb..
    return __builtin_shufflevector(q0, q1, 0, 1, 2, 3, 4, 5, 6, 7,
                                   8, 9, 10, 11, 12, 13, 14, 15);
}

// B-matrix fragment (32x16, K x N), B[k][n] = W[n][k].
// lane l: n = l&15; kb = (l<16)?0:8; element i: k = (i>>3)*16 + kb + (i&7).
// rowptr points at W_bf16[n_global][kk]; two packed 16B loads.
__device__ __forceinline__ v16bf load_b_frag(const __bf16* __restrict__ rowptr, int lane) {
    const int kb = (lane & 16) ? 8 : 0;
    v8bf q0 = *reinterpret_cast<const v8bf*>(rowptr + kb);
    v8bf q1 = *reinterpret_cast<const v8bf*>(rowptr + kb + 16);
    return __builtin_shufflevector(q0, q1, 0, 1, 2, 3, 4, 5, 6, 7,
                                   8, 9, 10, 11, 12, 13, 14, 15);
}

// ---------------------------------------------------------------------------
// K0: f32 -> bf16 bulk converter (4 elements/thread, vectorized).
// ---------------------------------------------------------------------------
__global__ void cvt_bf16_kernel(const float* __restrict__ src,
                                __bf16* __restrict__ dst) {
    const int i = blockIdx.x * blockDim.x + threadIdx.x;
    float4 q = reinterpret_cast<const float4*>(src)[i];
    v4bf o;
    o[0] = (__bf16)q.x; o[1] = (__bf16)q.y; o[2] = (__bf16)q.z; o[3] = (__bf16)q.w;
    reinterpret_cast<v4bf*>(dst)[i] = o;
}

// ---------------------------------------------------------------------------
// K1: per-graph scalar s_g = relu(g_feats[g]) . Wl[0:256] + bl
// ---------------------------------------------------------------------------
__global__ void graph_score_kernel(const float* __restrict__ gfeat,
                                   const float* __restrict__ Wl,
                                   const float* __restrict__ bl,
                                   float* __restrict__ s_g) {
    __shared__ float red[256];
    const int g = blockIdx.x, t = threadIdx.x;
    float gv = gfeat[(size_t)g * GFEAT + t];
    gv = gv > 0.0f ? gv : 0.0f;
    red[t] = gv * Wl[t];
    __syncthreads();
    for (int s = 128; s > 0; s >>= 1) {
        if (t < s) red[t] += red[t + s];
        __syncthreads();
    }
    if (t == 0) s_g[g] = red[0] + bl[0];
}

// ---------------------------------------------------------------------------
// K2: per-node logit z_v = leaky_relu(nf_v . Wl[256:512] + s_{seg(v)})
// One wave (32 lanes) per node; 2xfloat4 per lane -> coalesced 1KB rows.
// Streaming pass #1 over node_feats.
// ---------------------------------------------------------------------------
__global__ void node_logit_kernel(const float* __restrict__ nf,
                                  const float* __restrict__ Wl,
                                  const int*   __restrict__ seg,
                                  const float* __restrict__ s_g,
                                  float* __restrict__ z) {
    const int v    = blockIdx.x * 8 + (threadIdx.x >> 5);
    const int lane = threadIdx.x & 31;
    const float* row = nf + (size_t)v * NFEAT + lane * 8;
    const float* wn  = Wl + GFEAT + lane * 8;   // node-feature half of Wl
    float4 x0 = *reinterpret_cast<const float4*>(row);
    float4 x1 = *reinterpret_cast<const float4*>(row + 4);
    float4 w0 = *reinterpret_cast<const float4*>(wn);
    float4 w1 = *reinterpret_cast<const float4*>(wn + 4);
    float p = x0.x * w0.x + x0.y * w0.y + x0.z * w0.z + x0.w * w0.w +
              x1.x * w1.x + x1.y * w1.y + x1.z * w1.z + x1.w * w1.w;
    p += __shfl_xor(p, 16);
    p += __shfl_xor(p, 8);
    p += __shfl_xor(p, 4);
    p += __shfl_xor(p, 2);
    p += __shfl_xor(p, 1);
    if (lane == 0) {
        float zz = p + s_g[seg[v]];
        z[v] = zz > 0.0f ? zz : 0.01f * zz;
    }
}

// ---------------------------------------------------------------------------
// K3: fused segmented softmax + weighted feature sum -> bf16 wsum.
// Block per graph; binary search on sorted segment_ids. Each of 256 threads
// owns one feature column; all threads walk nodes together. Streaming pass #2
// over node_feats, pipelined depth-4 through LDS with ASYNC DMA when
// available. The exp-weight denominator is replicated per thread (no second
// reduction needed).
// ---------------------------------------------------------------------------
__global__ void attn_pool_kernel(const float* __restrict__ nf,
                                 const int*   __restrict__ seg,
                                 const float* __restrict__ z,
                                 __bf16* __restrict__ wsum_bf) {
    const int g = blockIdx.x, t = threadIdx.x;
    // lower_bound(seg, g) / lower_bound(seg, g+1), redundantly per thread
    int lo = 0, hi = NV;
    while (lo < hi) { int mid = (lo + hi) >> 1; if (seg[mid] < g) lo = mid + 1; else hi = mid; }
    const int start = lo;
    hi = NV;
    while (lo < hi) { int mid = (lo + hi) >> 1; if (seg[mid] < g + 1) lo = mid + 1; else hi = mid; }
    const int end = lo;

    __shared__ float red[256];
    float lm = -3.402823e38f;
    for (int v = start + t; v < end; v += 256) lm = fmaxf(lm, z[v]);
    red[t] = lm;
    __syncthreads();
    for (int s = 128; s > 0; s >>= 1) {
        if (t < s) red[t] = fmaxf(red[t], red[t + s]);
        __syncthreads();
    }
    const float m = red[0];

    float acc = 0.0f, dsum = 0.0f;
#if USE_ASYNC_LDS
    // Depth-4 ring of rows in LDS; each thread DMAs + consumes its own column,
    // so per-wave ASYNCcnt ordering is sufficient (no barriers needed).
    // Out-of-range iterations issue padding loads (row 0 -> dummy) so exactly
    // one async op is issued per iteration and `s_wait_asynccnt 3` always
    // guarantees the oldest transfer has landed. The diagnostic-confirmed
    // builtin signature is (AS1 int* src, AS3 int* lds_dst, imm off, imm cpol).
    __shared__ float stage[4][256];
    __shared__ float dummy[256];
    typedef __attribute__((address_space(1))) int gint;
    typedef __attribute__((address_space(3))) int lint;
#pragma unroll
    for (int d = 0; d < 4; ++d) {
        const int vv = start + d;
        const float* gp = nf + (size_t)(vv < end ? vv : 0) * NFEAT + t;
        float* lp = (vv < end) ? &stage[d][t] : &dummy[t];
        __builtin_amdgcn_global_load_async_to_lds_b32(
            (gint*)(void*)gp, (lint*)(void*)lp, 0, 0);
    }
    int slot = 0;
    for (int v = start; v < end; ++v) {
        __builtin_amdgcn_s_wait_asynccnt(3);
        const float e = __expf(z[v] - m);
        dsum += e;
        acc += e * stage[slot][t];   // LDS read completes (dscnt) before next issue
        const int vn = v + 4;
        const float* gp = nf + (size_t)(vn < end ? vn : 0) * NFEAT + t;
        float* lp = (vn < end) ? &stage[slot][t] : &dummy[t];
        __builtin_amdgcn_global_load_async_to_lds_b32(
            (gint*)(void*)gp, (lint*)(void*)lp, 0, 0);
        slot = (slot + 1) & 3;
    }
#else
    for (int v = start; v < end; ++v) {
        __builtin_prefetch(nf + (size_t)(v + 3) * NFEAT + t, 0, 0);
        const float e = __expf(z[v] - m);
        dsum += e;
        acc += e * nf[(size_t)v * NFEAT + t];
    }
#endif
    wsum_bf[(size_t)g * NFEAT + t] = (__bf16)((end > start) ? (acc / dsum) : 0.0f);
}

// ---------------------------------------------------------------------------
// K4: context = elu(wsum @ Wp.T + bp) -> bf16   [4096 x 256] via WMMA bf16.
// One 16x16 tile per wave; K-loop of 8 x v_wmma_f32_16x16x32_bf16 with
// packed-bf16 fragment loads (no conversions in the loop).
// ---------------------------------------------------------------------------
__global__ void ctx_project_kernel(const __bf16* __restrict__ wsum_bf,
                                   const __bf16* __restrict__ Wp_bf,
                                   const float*  __restrict__ bp,
                                   __bf16* __restrict__ ctx_bf) {
    const int wave = blockIdx.x * (blockDim.x >> 5) + (threadIdx.x >> 5);
    const int lane = threadIdx.x & 31;
    const int mt = wave / (GFEAT / 16);
    const int nt = wave % (GFEAT / 16);
    const int mrow = mt * 16 + (lane & 15);
    const int ncol = nt * 16 + (lane & 15);

    v8f c = {};
    for (int kk = 0; kk < NFEAT; kk += 32) {
        v16bf a = load_a_frag(wsum_bf + (size_t)mrow * NFEAT + kk, lane);
        v16bf b = load_b_frag(Wp_bf   + (size_t)ncol * NFEAT + kk, lane);
        c = wmma_bf16(a, b, c);
    }
    const float bpv = bp[ncol];
    const int rbase = mt * 16 + ((lane & 16) ? 8 : 0);
#pragma unroll
    for (int r = 0; r < 8; ++r) {
        float v = c[r] + bpv;
        v = v > 0.0f ? v : (__expf(v) - 1.0f);   // elu
        ctx_bf[(size_t)(rbase + r) * GFEAT + ncol] = (__bf16)v;
    }
}

// ---------------------------------------------------------------------------
// K5: fused GRUCell. Six WMMA accumulators per 16x16 tile of h_new
// (context/g_feats x {r,z,n} slices of W_ih.T / W_hh.T), gate math in
// registers, direct store of h_new.
// ---------------------------------------------------------------------------
__global__ void gru_kernel(const __bf16* __restrict__ ctx_bf,
                           const __bf16* __restrict__ gf_bf,
                           const float*  __restrict__ gfeat,
                           const __bf16* __restrict__ Wih_bf,
                           const __bf16* __restrict__ Whh_bf,
                           const float*  __restrict__ bih,
                           const float*  __restrict__ bhh,
                           float* __restrict__ out) {
    const int wave = blockIdx.x * (blockDim.x >> 5) + (threadIdx.x >> 5);
    const int lane = threadIdx.x & 31;
    const int mt = wave / (GFEAT / 16);
    const int nt = wave % (GFEAT / 16);
    const int mrow = mt * 16 + (lane & 15);
    const int ncol = nt * 16 + (lane & 15);

    v8f cir = {}, ciz = {}, cin = {}, chr_ = {}, chz = {}, chn = {};
    for (int kk = 0; kk < GFEAT; kk += 32) {
        v16bf aC = load_a_frag(ctx_bf + (size_t)mrow * GFEAT + kk, lane);
        v16bf aH = load_a_frag(gf_bf  + (size_t)mrow * GFEAT + kk, lane);
        v16bf br = load_b_frag(Wih_bf + (size_t)(ncol            ) * GFEAT + kk, lane);
        v16bf bz = load_b_frag(Wih_bf + (size_t)(ncol +     GFEAT) * GFEAT + kk, lane);
        v16bf bn = load_b_frag(Wih_bf + (size_t)(ncol + 2 * GFEAT) * GFEAT + kk, lane);
        cir = wmma_bf16(aC, br, cir);
        ciz = wmma_bf16(aC, bz, ciz);
        cin = wmma_bf16(aC, bn, cin);
        v16bf hr = load_b_frag(Whh_bf + (size_t)(ncol            ) * GFEAT + kk, lane);
        v16bf hz = load_b_frag(Whh_bf + (size_t)(ncol +     GFEAT) * GFEAT + kk, lane);
        v16bf hn = load_b_frag(Whh_bf + (size_t)(ncol + 2 * GFEAT) * GFEAT + kk, lane);
        chr_ = wmma_bf16(aH, hr, chr_);
        chz  = wmma_bf16(aH, hz, chz);
        chn  = wmma_bf16(aH, hn, chn);
    }

    const float bir = bih[ncol],             bhr = bhh[ncol];
    const float biz = bih[ncol + GFEAT],     bhz = bhh[ncol + GFEAT];
    const float bin = bih[ncol + 2 * GFEAT], bhn = bhh[ncol + 2 * GFEAT];
    const int rbase = mt * 16 + ((lane & 16) ? 8 : 0);
#pragma unroll
    for (int r = 0; r < 8; ++r) {
        const float rg = sigmoidf_((cir[r] + bir) + (chr_[r] + bhr));
        const float ug = sigmoidf_((ciz[r] + biz) + (chz[r]  + bhz));
        const float ng = tanhf((cin[r] + bin) + rg * (chn[r] + bhn));
        const float hprev = gfeat[(size_t)(rbase + r) * GFEAT + ncol];
        out[(size_t)(rbase + r) * GFEAT + ncol] = (1.0f - ug) * ng + ug * hprev;
    }
}

// ---------------------------------------------------------------------------
extern "C" void kernel_launch(void* const* d_in, const int* in_sizes, int n_in,
                              void* d_out, int out_size, void* d_ws, size_t ws_size,
                              hipStream_t stream) {
    const float* nf    = (const float*)d_in[0];   // [V, NF]
    const float* gfeat = (const float*)d_in[1];   // [G, GF]
    const int*   seg   = (const int*)  d_in[2];   // [V]
    const float* Wl    = (const float*)d_in[3];   // [1, NF+GF]
    const float* bl    = (const float*)d_in[4];   // [1]
    const float* Wp    = (const float*)d_in[5];   // [GF, NF]
    const float* bp    = (const float*)d_in[6];   // [GF]
    const float* Wih   = (const float*)d_in[7];   // [3GF, GF]
    const float* Whh   = (const float*)d_in[8];   // [3GF, GF]
    const float* bih   = (const float*)d_in[9];   // [3GF]
    const float* bhh   = (const float*)d_in[10];  // [3GF]
    float* out = (float*)d_out;                   // [G, GF]

    // scratch layout: f32 regions then bf16 regions, all 16B-aligned
    float* ws   = (float*)d_ws;
    float* s_g  = ws;                                   // [G]
    float* z    = s_g + NG;                             // [V]
    __bf16* bfbase  = (__bf16*)(z + NV);
    __bf16* wsum_bf = bfbase;                           // [G, NF]
    __bf16* ctx_bf  = wsum_bf + (size_t)NG * NFEAT;     // [G, GF]
    __bf16* gf_bf   = ctx_bf  + (size_t)NG * GFEAT;     // [G, GF]
    __bf16* Wp_bf   = gf_bf   + (size_t)NG * GFEAT;     // [GF, NF]
    __bf16* Wih_bf  = Wp_bf   + (size_t)GFEAT * NFEAT;  // [3GF, GF]
    __bf16* Whh_bf  = Wih_bf  + (size_t)3 * GFEAT * GFEAT;

    // one-time bf16 conversion of all GEMM operands (L2-resident, tiny)
    cvt_bf16_kernel<<<(GFEAT * NFEAT) / 1024, 256, 0, stream>>>(Wp, Wp_bf);
    cvt_bf16_kernel<<<(3 * GFEAT * GFEAT) / 1024, 256, 0, stream>>>(Wih, Wih_bf);
    cvt_bf16_kernel<<<(3 * GFEAT * GFEAT) / 1024, 256, 0, stream>>>(Whh, Whh_bf);
    cvt_bf16_kernel<<<(NG * GFEAT) / 1024, 256, 0, stream>>>(gfeat, gf_bf);

    graph_score_kernel<<<NG, 256, 0, stream>>>(gfeat, Wl, bl, s_g);
    node_logit_kernel<<<NV / 8, 256, 0, stream>>>(nf, Wl, seg, s_g, z);
    attn_pool_kernel<<<NG, 256, 0, stream>>>(nf, seg, z, wsum_bf);
    // 4096 tiles, 8 waves per 256-thread block
    ctx_project_kernel<<<(NG / 16) * (GFEAT / 16) / 8, 256, 0, stream>>>(wsum_bf, Wp_bf, bp, ctx_bf);
    gru_kernel<<<(NG / 16) * (GFEAT / 16) / 8, 256, 0, stream>>>(ctx_bf, gf_bf, gfeat, Wih_bf, Whh_bf, bih, bhh, out);
}
